// Quantizer_module_4157528342985
// MI455X (gfx1250) — compile-verified, matrix operand order checked by reading, not run on hardware
//
#include <hip/hip_runtime.h>
#include <hip/hip_bf16.h>
#include <stdint.h>

// ---------------------------------------------------------------------------
// VQ quantizer for MI455X (gfx1250, wave32, WMMA + TDM).
//   z_q[n,:] = emb[argmin_k(||x_n - e_k||^2), :]
// argmin_k(||x||^2 + ||e||^2 - 2<x,e>) == argmin_k(||e||^2 - 2<x,e>)
// GEMM via v_wmma_f32_16x16x32_bf16 (f32 accumulate); B tiles staged to LDS
// by the Tensor Data Mover (double-buffered, overlapped with compute) and
// shared by all 8 waves; all 8 B chunks loaded into distinct register octets
// before the 8 back-to-back WMMAs so ds-load latency is not serialized with
// the matrix pipeline.
// ---------------------------------------------------------------------------

typedef __attribute__((ext_vector_type(16))) __bf16 v16bf;
typedef __attribute__((ext_vector_type(8)))  float  v8f;
typedef __attribute__((ext_vector_type(4)))  unsigned v4u;
typedef __attribute__((ext_vector_type(8)))  int      v8i_t;
typedef __attribute__((ext_vector_type(4)))  int      v4i_t;

#define N_E    1024
#define E_DIM  256
#define N_TOK  262144
#define NTILES (N_E / 16)     // 64 column tiles of 16 embeddings
#define KCHUNK (E_DIM / 32)   // 8 K-chunks of 32
#define BLOCK_M 128           // tokens per block
#define THREADS 256           // 8 waves
#define TILE_BYTES 8192       // one packed B tile: 16 cols x 256 K x bf16

// LDS layout (dynamic shared)
#define XS_U32_STRIDE 132                         // 264 bf16, pad vs bank conflicts
#define XS_BYTES  (BLOCK_M * XS_U32_STRIDE * 4)   // 67584
#define BBUF_OFF  XS_BYTES                        // 2 x 8KB double buffer
#define IDX_OFF   (BBUF_OFF + 2 * TILE_BYTES)
#define SMEM_BYTES (IDX_OFF + BLOCK_M * 4)        // 84480 total

#if defined(__has_builtin)
#if __has_builtin(__builtin_amdgcn_tensor_load_to_lds) && \
    __has_builtin(__builtin_amdgcn_s_wait_tensorcnt)
#define HAVE_TDM 1
#endif
#endif
#ifndef HAVE_TDM
#define HAVE_TDM 0
#endif

// round-to-nearest-even f32 -> bf16 bits
__device__ __forceinline__ unsigned short f2bf(float f) {
  unsigned u = __float_as_uint(f);
  unsigned r = 0x7fffu + ((u >> 16) & 1u);
  return (unsigned short)((u + r) >> 16);
}

// Even K base of bf16-pair v (0..7) for lane half `hi`, per the CDNA5 16-bit
// A/B matrix VGPR layout (CDNA5 ISA).
__device__ __forceinline__ int kpair(int hi, int v) {
  return 2 * (v & 3) + ((v >= 4) ? 16 : 0) + (hi ? 8 : 0);
}

#if HAVE_TDM
// Issue a TDM copy of one contiguous TILE_BYTES block (1024 x 8B elements,
// 2D degenerate tile) from global memory into LDS at byte offset lds_off.
// 6-arg builtin: (uint32x4 g0, int32x8 g1, int32x4 g2, int32x4 g3, int32x8, i32 cpol)
__device__ __forceinline__ void tdm_load_tile(unsigned lds_off, const void* g) {
  unsigned long long ga = (unsigned long long)g;
  v4u g0;
  g0[0] = 1u;                                   // count=1, user descriptor
  g0[1] = lds_off;                              // D#.lds_addr
  g0[2] = (unsigned)(ga & 0xffffffffu);         // D#.global_addr[31:0]
  g0[3] = (unsigned)(ga >> 32) | (2u << 30);    // addr[56:32] | type=2
  v8i_t g1;
  g1[0] = (int)(3u << 16);                      // data_size=3 (8 bytes)
  g1[1] = (int)(1024u << 16);                   // tensor_dim0 = 1024 (lo16)
  g1[2] = (int)(1u << 16);                      // tensor_dim1 = 1
  g1[3] = (int)(1024u << 16);                   // tile_dim0 = 1024
  g1[4] = 1;                                    // tile_dim1 = 1
  g1[5] = 1024;                                 // tensor_dim0_stride = 1024
  g1[6] = 0;
  g1[7] = 0;
  v4i_t z4 = {0, 0, 0, 0};
  v8i_t z8 = {0, 0, 0, 0, 0, 0, 0, 0};
  __builtin_amdgcn_tensor_load_to_lds(g0, g1, z4, z4, z8, 0);
}
#endif

// ---------------------------------------------------------------------------
// Kernel 1: per-row ||e_k||^2 (exact fp32) + pre-shuffle emb into bf16 packed
// B-matrix lane layout: pack[((ntile*8+chunk)*32 + lane)*16 + j]
// Each 16-column tile is a contiguous 8KB block -> TDM-friendly.
// ---------------------------------------------------------------------------
__global__ void __launch_bounds__(256)
vq_pack_kernel(const float* __restrict__ emb,
               unsigned short* __restrict__ pack,
               float* __restrict__ enorm) {
  __shared__ float red[256];
  const int row = blockIdx.x;     // 0..1023
  const int t   = threadIdx.x;    // 0..255

  float v = emb[row * E_DIM + t];
  red[t] = v * v;
  __syncthreads();
  for (int s = 128; s > 0; s >>= 1) {
    if (t < s) red[t] += red[t + s];
    __syncthreads();
  }
  if (t == 0) enorm[row] = red[0];

  const int p     = blockIdx.x * 256 + t;   // packed element index
  const int j     = p & 15;
  const int lane  = (p >> 4) & 31;
  const int chunk = (p >> 9) & 7;
  const int ntile = p >> 12;
  const int src_row = ntile * 16 + (lane & 15);
  const int src_col = chunk * 32 + kpair(lane >> 4, j >> 1) + (j & 1);
  pack[p] = f2bf(emb[src_row * E_DIM + src_col]);
}

// ---------------------------------------------------------------------------
// Kernel 2: main search + gather.
// ---------------------------------------------------------------------------
__global__ void __launch_bounds__(THREADS)
vq_search_kernel(const float* __restrict__ x,
                 const float* __restrict__ emb,
                 const unsigned short* __restrict__ pack,
                 const float* __restrict__ enorm,
                 float* __restrict__ zq,
                 float* __restrict__ idx_out) {
  extern __shared__ char smem[];
  unsigned int* xsu   = (unsigned int*)smem;            // packed bf16 x tile
  char*         bbuf  = smem + BBUF_OFF;                // B double buffer
  int*          idxsh = (int*)(smem + IDX_OFF);

  const int t    = threadIdx.x;
  const int lane = t & 31;
  const int wave = t >> 5;
  const int hi   = lane >> 4;
  const long long m0 = (long long)blockIdx.x * BLOCK_M;

  // ---- stage x tile -> LDS as packed bf16 (coalesced float2 loads) ----
  for (int rr = 0; rr < BLOCK_M; rr += 2) {
    const int row = rr + (t >> 7);
    const int c2  = (t & 127);
    const float2 xv = *(const float2*)(x + (m0 + row) * E_DIM + c2 * 2);
    xsu[row * XS_U32_STRIDE + c2] =
        (unsigned)f2bf(xv.x) | ((unsigned)f2bf(xv.y) << 16);
  }
  __syncthreads();

  // ---- this wave's 16x256 A tile into registers (8 x v16bf) ----
  union AV { v16bf v; unsigned u[8]; };
  AV A[KCHUNK];
  const int mrow = wave * 16 + (lane & 15);
  for (int c = 0; c < KCHUNK; ++c)
    for (int v = 0; v < 8; ++v)
      A[c].u[v] = xsu[mrow * XS_U32_STRIDE + c * 16 + (kpair(hi, v) >> 1)];

  float bestd[8];
  int   besti[8];
  for (int r = 0; r < 8; ++r) { bestd[r] = 3.4e38f; besti[r] = 0; }

#if HAVE_TDM
  // prologue: TDM tile 0 into buffer 0
  if (wave == 0) {
    tdm_load_tile((unsigned)(unsigned long long)bbuf, pack);
    __builtin_amdgcn_s_wait_tensorcnt(0);
  }
  __syncthreads();
#endif

  for (int nt = 0; nt < NTILES; ++nt) {
#if HAVE_TDM
    const char* btile = bbuf + (nt & 1) * TILE_BYTES;
    if (nt + 1 < NTILES && wave == 0) {
      // overlap next tile's TDM copy with this tile's 8 WMMAs
      tdm_load_tile((unsigned)(unsigned long long)(bbuf + ((nt + 1) & 1) * TILE_BYTES),
                    (const char*)pack + (nt + 1) * TILE_BYTES);
    }
#else
    // fallback: cooperative synchronous copy, single buffer
    __syncthreads();
    {
      const uint4* g = (const uint4*)((const char*)pack + nt * TILE_BYTES);
      uint4* s = (uint4*)bbuf;
      s[t * 2]     = g[t * 2];
      s[t * 2 + 1] = g[t * 2 + 1];
    }
    __syncthreads();
    const char* btile = bbuf;
#endif

    // ---- load ALL 8 B chunks into distinct register octets first (16
    //      clustered ds_load_b128), then issue the 8 WMMAs back-to-back;
    //      in-order DScnt lets the scheduler overlap load latency with
    //      the matrix pipeline instead of wait-0 before each WMMA. ----
    union BV { v16bf v; uint4 q[2]; };
    BV B[KCHUNK];
#pragma unroll
    for (int c = 0; c < KCHUNK; ++c) {
      const uint4* s = (const uint4*)(btile + (c * 32 + lane) * 32);
      B[c].q[0] = s[0];
      B[c].q[1] = s[1];
    }
    v8f acc = {0.f, 0.f, 0.f, 0.f, 0.f, 0.f, 0.f, 0.f};
#pragma unroll
    for (int c = 0; c < KCHUNK; ++c) {
      acc = __builtin_amdgcn_wmma_f32_16x16x32_bf16(
          /*neg_a=*/false, A[c].v, /*neg_b=*/false, B[c].v,
          /*c_mod=*/(short)0, acc, /*reuse_a=*/false, /*reuse_b=*/false);
    }

    const int   n  = nt * 16 + (lane & 15);   // this lane's column
    const float en = enorm[n];
    for (int r = 0; r < 8; ++r) {
      const float d = en - 2.0f * acc[r];
      if (d < bestd[r]) { bestd[r] = d; besti[r] = n; }
    }

#if HAVE_TDM
    if (nt + 1 < NTILES) {
      if (wave == 0) __builtin_amdgcn_s_wait_tensorcnt(0);
      __syncthreads();   // publish next B tile to all waves
    }
#endif
  }

  // ---- reduce (min,argmin) across the 16 lanes of each half-wave ----
  for (int r = 0; r < 8; ++r) {
    float d  = bestd[r];
    int   bi = besti[r];
    for (int m = 1; m < 16; m <<= 1) {
      const float od = __shfl_xor(d, m, 32);
      const int   oi = __shfl_xor(bi, m, 32);
      if (od < d || (od == d && oi < bi)) { d = od; bi = oi; }
    }
    bestd[r] = d;
    besti[r] = bi;
  }
  if ((lane & 15) == 0) {
    for (int r = 0; r < 8; ++r)
      idxsh[wave * 16 + hi * 8 + r] = besti[r];
  }
  __syncthreads();

  // ---- emit indices (as float, concatenated after z_q) ----
  if (t < BLOCK_M) idx_out[m0 + t] = (float)idxsh[t];

  // ---- gather z_q rows (coalesced, emb rows are L2-resident) ----
  for (int r = 0; r < BLOCK_M; ++r) {
    const int k = idxsh[r];
    zq[(m0 + r) * E_DIM + t] = emb[(long long)k * E_DIM + t];
  }
}

// ---------------------------------------------------------------------------
extern "C" void kernel_launch(void* const* d_in, const int* in_sizes, int n_in,
                              void* d_out, int out_size, void* d_ws, size_t ws_size,
                              hipStream_t stream) {
  const float* x   = (const float*)d_in[0];   // [N_TOK, E_DIM] fp32
  const float* emb = (const float*)d_in[1];   // [N_E,  E_DIM] fp32

  float*          enorm = (float*)d_ws;                            // 4 KB
  unsigned short* pack  = (unsigned short*)((char*)d_ws + 4096);   // 512 KB

  float* zq      = (float*)d_out;                         // [N_TOK, E_DIM]
  float* idx_out = (float*)d_out + (size_t)N_TOK * E_DIM; // [N_TOK]

  vq_pack_kernel<<<N_E, 256, 0, stream>>>(emb, pack, enorm);
  vq_search_kernel<<<N_TOK / BLOCK_M, THREADS, SMEM_BYTES, stream>>>(
      x, emb, pack, enorm, zq, idx_out);
}
